// GraphMAE_82016695484546
// MI455X (gfx1250) — matrix operand order repaired and checked by reading, not compiled, not run on hardware
//
#include <hip/hip_runtime.h>
#include <hip/hip_bf16.h>

// ---------------------------------------------------------------------------
// GraphMAE GIN forward for gfx1250 (MI455X).
// GEMMs: v_wmma_f32_16x16x32_bf16, f32 accum. Weights pre-converted to a
// transposed, K-padded bf16 layout so the K-loop is branchless; activations
// converted f32->bf16 on the fly with vectorized loads. Scatter = native
// global_atomic_add_f32.
// ---------------------------------------------------------------------------

typedef __attribute__((ext_vector_type(16))) __bf16        v16bf;
typedef __attribute__((ext_vector_type(8)))  float         v8f;
typedef __attribute__((ext_vector_type(4)))  unsigned int  u32x4;

union BF16Frag { u32x4 u[2]; v16bf v; };

#define GIN_N   200000
#define GIN_E   400000
#define GIN_D   300
#define GIN_H   600
#define GIN_L   5
#define GIN_OUT 119

#define TILE_M 128
#define TILE_N 128
#define TILE_K 32
#define LDS_K  40   // bf16 elems per LDS row (32 + 8 pad); 80B stride keeps 16B align

__device__ __forceinline__ unsigned int pack_bf16x2(float a, float b) {
    return (__float_as_uint(a) >> 16) | (__float_as_uint(b) & 0xffff0000u);
}

// ------------------------------ elementwise --------------------------------
// One block per node/edge, threadIdx.x = feature dim (blockDim = 320 >= 300).

__global__ void embed_kernel(const int* __restrict__ x,
                             const float* __restrict__ emb1,
                             const float* __restrict__ emb2,
                             float* __restrict__ h) {
    int n = blockIdx.x, d = threadIdx.x;
    if (d >= GIN_D) return;
    h[(size_t)n * GIN_D + d] = emb1[(size_t)x[2 * n] * GIN_D + d]
                             + emb2[(size_t)x[2 * n + 1] * GIN_D + d];
}

// agg = h + (eemb1[4] + eemb2[0])   (self-loop message)
__global__ void init_agg_kernel(const float* __restrict__ h,
                                const float* __restrict__ ee1,
                                const float* __restrict__ ee2,
                                float* __restrict__ agg) {
    int n = blockIdx.x, d = threadIdx.x;
    if (d >= GIN_D) return;
    size_t i = (size_t)n * GIN_D + d;
    agg[i] = h[i] + ee1[4 * GIN_D + d] + ee2[d];
}

// agg[dst] += h[src] + eemb1[bond_type] + eemb2[bond_dir]
__global__ void edge_scatter_kernel(const float* __restrict__ h,
                                    const int* __restrict__ eidx,
                                    const int* __restrict__ eattr,
                                    const float* __restrict__ ee1,
                                    const float* __restrict__ ee2,
                                    float* __restrict__ agg) {
    int e = blockIdx.x, d = threadIdx.x;
    if (d >= GIN_D) return;
    int s  = eidx[e];
    int t  = eidx[GIN_E + e];
    int a0 = eattr[2 * e];
    int a1 = eattr[2 * e + 1];
    float msg = h[(size_t)s * GIN_D + d]
              + ee1[(size_t)a0 * GIN_D + d]
              + ee2[(size_t)a1 * GIN_D + d];
    unsafeAtomicAdd(&agg[(size_t)t * GIN_D + d], msg);   // global_atomic_add_f32
}

__global__ void prelu_kernel(const float* __restrict__ h,
                             const float* __restrict__ alpha,
                             float* __restrict__ p, int total) {
    int i = blockIdx.x * blockDim.x + threadIdx.x;
    if (i >= total) return;
    float a = alpha[0];
    float v = h[i];
    p[i] = v >= 0.f ? v : a * v;
}

// W[K,Nc] f32 -> Wt[Nc,Kp] bf16, zero-padded in K (Kp = ceil32(K)).
__global__ void conv_weightsT_kernel(const float* __restrict__ W,
                                     unsigned short* __restrict__ Wt,
                                     int K, int Nc, int Kp) {
    int i = blockIdx.x * blockDim.x + threadIdx.x;
    if (i >= Nc * Kp) return;
    int n = i / Kp, k = i - n * Kp;
    float v = (k < K) ? W[(size_t)k * Nc + n] : 0.f;
    Wt[i] = (unsigned short)(__float_as_uint(v) >> 16);
}

// ------------------------------ WMMA GEMM ----------------------------------
// C[M,Nc] = epilogue(A[M,K] @ B[K,Nc]),  B given as Wt[Nc,Kp] bf16 (padded).
// epilogue: (+bias) -> (BN eval) -> (ReLU) -> (zero masked rows)
// Block tile 128x128, 8 waves: 2 along M (4 subtiles) x 4 along N (2 subtiles).
// K-loop is fully branchless: B's K-pad is zero, A's row index is clamped.

__global__ __launch_bounds__(256)
void wmma_gemm_kernel(const float* __restrict__ A,
                      const unsigned short* __restrict__ Bt,
                      const float* __restrict__ bias,
                      const float* __restrict__ bnG, const float* __restrict__ bnB,
                      const float* __restrict__ bnM, const float* __restrict__ bnV,
                      const unsigned char* __restrict__ maskRows,
                      float* __restrict__ C,
                      int M, int K, int Nc, int Kp, int doRelu) {
    __shared__ __align__(16) unsigned short sA[TILE_M * LDS_K];   // [m][k] bf16
    __shared__ __align__(16) unsigned short sB[TILE_N * LDS_K];   // [n][k] bf16

    const int tid  = threadIdx.x;
    const int lane = tid & 31;
    const int wave = tid >> 5;
    const int hlf  = lane >> 4;
    const int l16  = lane & 15;

    const int blockM = blockIdx.y * TILE_M;
    const int blockN = blockIdx.x * TILE_N;
    const int waveM  = (wave & 1) * 64;
    const int waveN  = (wave >> 1) * 32;

    // --- staging assignments (branchless by clamping) ---
    const int aRowL  = tid >> 1;                         // 0..127
    const int aCb    = (tid & 1) * 16;                   // 0 / 16
    const int aRowG  = min(blockM + aRowL, M - 1);
    const float* Arow = A + (size_t)aRowG * K + aCb;     // + k0 at use
    unsigned short* sAdst = &sA[aRowL * LDS_K + aCb];

    const int bNL    = tid & 127;                        // 0..127
    const int bKb    = (tid >> 7) * 16;                  // 0 / 16
    const int bColG  = min(blockN + bNL, Nc - 1);
    const unsigned short* Brow = Bt + (size_t)bColG * Kp + bKb;
    unsigned short* sBdst = &sB[bNL * LDS_K + bKb];

    v8f acc[4][2];
#pragma unroll
    for (int mi = 0; mi < 4; ++mi)
#pragma unroll
        for (int ni = 0; ni < 2; ++ni)
#pragma unroll
            for (int r = 0; r < 8; ++r) acc[mi][ni][r] = 0.f;

    for (int k0 = 0; k0 < Kp; k0 += TILE_K) {
        // --- stage A: 4x aligned float4 loads, pack to bf16, 2x ds_store_b128
        {
            const float4* ap = (const float4*)(Arow + k0);
            float4 f0 = ap[0], f1 = ap[1], f2 = ap[2], f3 = ap[3];
            u32x4 lo, hi;
            lo.x = pack_bf16x2(f0.x, f0.y); lo.y = pack_bf16x2(f0.z, f0.w);
            lo.z = pack_bf16x2(f1.x, f1.y); lo.w = pack_bf16x2(f1.z, f1.w);
            hi.x = pack_bf16x2(f2.x, f2.y); hi.y = pack_bf16x2(f2.z, f2.w);
            hi.z = pack_bf16x2(f3.x, f3.y); hi.w = pack_bf16x2(f3.z, f3.w);
            ((u32x4*)sAdst)[0] = lo;
            ((u32x4*)sAdst)[1] = hi;
        }
        // --- stage B: 2x b128 loads from pre-converted bf16, 2x ds_store_b128
        {
            const u32x4* bp = (const u32x4*)(Brow + k0);
            u32x4 b0 = bp[0], b1 = bp[1];
            ((u32x4*)sBdst)[0] = b0;
            ((u32x4*)sBdst)[1] = b1;
        }
        if (k0 + TILE_K < Kp)
            __builtin_prefetch(Arow + k0 + TILE_K, 0, 1);   // global_prefetch_b8
        __syncthreads();

        // --- fragments per ISA wave32 16-bit layouts ---
        v16bf afrag[4];
#pragma unroll
        for (int mi = 0; mi < 4; ++mi) {
            int r = waveM + mi * 16 + l16;
            BF16Frag f;
            f.u[0] = *(const u32x4*)&sA[r * LDS_K + hlf * 8];
            f.u[1] = *(const u32x4*)&sA[r * LDS_K + 16 + hlf * 8];
            afrag[mi] = f.v;
        }
        v16bf bfrag[2];
#pragma unroll
        for (int ni = 0; ni < 2; ++ni) {
            int c = waveN + ni * 16 + l16;
            BF16Frag f;
            f.u[0] = *(const u32x4*)&sB[c * LDS_K + hlf * 8];
            f.u[1] = *(const u32x4*)&sB[c * LDS_K + 16 + hlf * 8];
            bfrag[ni] = f.v;
        }
#pragma unroll
        for (int mi = 0; mi < 4; ++mi)
#pragma unroll
            for (int ni = 0; ni < 2; ++ni)
                acc[mi][ni] = __builtin_amdgcn_wmma_f32_16x16x32_bf16(
                    false, afrag[mi], false, bfrag[ni],
                    (short)0, acc[mi][ni], false, false);

        __syncthreads();
    }

    // --- fused epilogue ---
#pragma unroll
    for (int mi = 0; mi < 4; ++mi) {
#pragma unroll
        for (int ni = 0; ni < 2; ++ni) {
            int col = blockN + waveN + ni * 16 + l16;
            if (col >= Nc) continue;
            float bc = bias ? bias[col] : 0.f;
            float g = 1.f, bt = 0.f, mn = 0.f, inv = 1.f;
            if (bnG) {
                g   = bnG[col]; bt = bnB[col]; mn = bnM[col];
                inv = rsqrtf(bnV[col] + 1e-5f);
            }
#pragma unroll
            for (int r = 0; r < 8; ++r) {
                int row = blockM + waveM + mi * 16 + hlf * 8 + r;
                if (row >= M) continue;
                float v = acc[mi][ni][r] + bc;
                if (bnG) v = (v - mn) * inv * g + bt;
                if (doRelu) v = v > 0.f ? v : 0.f;
                if (maskRows && maskRows[row]) v = 0.f;
                C[(size_t)row * Nc + col] = v;
            }
        }
    }
}

// ------------------------------- launcher ----------------------------------

extern "C" void kernel_launch(void* const* d_in, const int* in_sizes, int n_in,
                              void* d_out, int out_size, void* d_ws, size_t ws_size,
                              hipStream_t stream) {
    const int*   x     = (const int*)d_in[0];
    const int*   eidx  = (const int*)d_in[1];
    const int*   eattr = (const int*)d_in[2];
    const unsigned char* mask = (const unsigned char*)d_in[3];
    const float* aemb1 = (const float*)d_in[4];
    const float* aemb2 = (const float*)d_in[5];
    const float* encW1 = (const float*)d_in[6];
    const float* encB1 = (const float*)d_in[7];
    const float* encW2 = (const float*)d_in[8];
    const float* encB2 = (const float*)d_in[9];
    const float* encE1 = (const float*)d_in[10];
    const float* encE2 = (const float*)d_in[11];
    const float* bnG   = (const float*)d_in[12];
    const float* bnB   = (const float*)d_in[13];
    const float* bnM   = (const float*)d_in[14];
    const float* bnV   = (const float*)d_in[15];
    const float* alpha = (const float*)d_in[16];
    const float* e2dW  = (const float*)d_in[17];
    const float* decW1 = (const float*)d_in[18];
    const float* decB1 = (const float*)d_in[19];
    const float* decW2 = (const float*)d_in[20];
    const float* decB2 = (const float*)d_in[21];
    const float* decE1 = (const float*)d_in[22];
    const float* decE2 = (const float*)d_in[23];

    float* h   = (float*)d_ws;                        // [N, D] f32
    float* agg = h   + (size_t)GIN_N * GIN_D;         // [N, D] f32
    float* hid = agg + (size_t)GIN_N * GIN_D;         // [N, H] f32
    unsigned short* wt = (unsigned short*)(hid + (size_t)GIN_N * GIN_H); // bf16 Wt scratch
    float* out = (float*)d_out;                       // [N, OUT] f32

    const int ND = GIN_N * GIN_D;
    dim3 featBlk(320);   // 10 waves, covers D=300

    auto gemm = [&](const float* A, const float* W, const float* bias,
                    const float* g, const float* bb, const float* mn, const float* vr,
                    const unsigned char* mrow, float* Cm, int M, int K, int Nc, int relu) {
        int Kp  = (K + 31) & ~31;
        int tot = Nc * Kp;
        conv_weightsT_kernel<<<(tot + 255) / 256, 256, 0, stream>>>(W, wt, K, Nc, Kp);
        dim3 grid((Nc + TILE_N - 1) / TILE_N, (M + TILE_M - 1) / TILE_M);
        wmma_gemm_kernel<<<grid, dim3(256), 0, stream>>>(
            A, wt, bias, g, bb, mn, vr, mrow, Cm, M, K, Nc, Kp, relu);
    };

    // atom embedding
    embed_kernel<<<dim3(GIN_N), featBlk, 0, stream>>>(x, aemb1, aemb2, h);

    // 5 encoder GINConv layers
    for (int l = 0; l < GIN_L; ++l) {
        const float* ee1 = encE1 + (size_t)l * 6 * GIN_D;
        const float* ee2 = encE2 + (size_t)l * 3 * GIN_D;
        init_agg_kernel<<<dim3(GIN_N), featBlk, 0, stream>>>(h, ee1, ee2, agg);
        edge_scatter_kernel<<<dim3(GIN_E), featBlk, 0, stream>>>(h, eidx, eattr, ee1, ee2, agg);
        // hid = relu(agg @ W1 + b1)
        gemm(agg, encW1 + (size_t)l * GIN_D * GIN_H, encB1 + (size_t)l * GIN_H,
             nullptr, nullptr, nullptr, nullptr, nullptr,
             hid, GIN_N, GIN_D, GIN_H, 1);
        // h = BN(hid @ W2 + b2); relu except last layer
        gemm(hid, encW2 + (size_t)l * GIN_H * GIN_D, encB2 + (size_t)l * GIN_D,
             bnG + (size_t)l * GIN_D, bnB + (size_t)l * GIN_D,
             bnM + (size_t)l * GIN_D, bnV + (size_t)l * GIN_D,
             nullptr, h, GIN_N, GIN_H, GIN_D, (l < GIN_L - 1) ? 1 : 0);
    }

    // encoder_to_decoder: h = (PReLU(h) @ e2d_W), zero masked rows (fused)
    prelu_kernel<<<(ND + 255) / 256, dim3(256), 0, stream>>>(h, alpha, agg, ND);
    gemm(agg, e2dW, nullptr, nullptr, nullptr, nullptr, nullptr, mask,
         h, GIN_N, GIN_D, GIN_D, 0);

    // decoder GINConv -> logits [N, 119]
    init_agg_kernel<<<dim3(GIN_N), featBlk, 0, stream>>>(h, decE1, decE2, agg);
    edge_scatter_kernel<<<dim3(GIN_E), featBlk, 0, stream>>>(h, eidx, eattr, decE1, decE2, agg);
    gemm(agg, decW1, decB1, nullptr, nullptr, nullptr, nullptr, nullptr,
         hid, GIN_N, GIN_D, GIN_H, 1);
    gemm(hid, decW2, decB2, nullptr, nullptr, nullptr, nullptr, nullptr,
         out, GIN_N, GIN_H, GIN_OUT, 0);
}
